// MultiScaleDeformableAttention_81724637708416
// MI455X (gfx1250) — matrix operand order repaired and checked by reading, not compiled, not run on hardware
//
#include <hip/hip_runtime.h>
#include <stdint.h>

// ---------------------------------------------------------------------------
// MultiScaleDeformableAttention for MI455X (gfx1250, wave32, WMMA bf16 + TDM)
// ---------------------------------------------------------------------------
// Pipeline:
//   [cvt]   query,value f32 -> bf16         (activations)
//   [trans] W_v, W_off, W_attn, W_out -> W^T bf16 (so B-fragments are K-contig)
//   [gemm]  v_proj = value_bf @ W_v^T + b_v            (f32 out)
//   [gemm]  off    = query_bf @ W_off^T + b_off        (f32 out)
//   [gemm]  attn_l = query_bf @ W_attn^T + b_attn      (f32 out)
//   [samp]  softmax over P, bilinear gather from v_proj, weighted sum
//   [gemm]  d_out  = out_i_bf @ W_out^T + b_out        (f32 out)
//
// GEMM: block = 8 waves = 256 M-rows x 64 N-cols. The 64x256 bf16 weight
// panel (32 KB) is DMA'd into LDS once per block by the Tensor Data Mover
// (fallback: cooperative load), then every wave reads B fragments with
// ds_load_b128. Each wave computes 32x64 (8 accumulators) so each A/B
// fragment is reused 4x/2x: ~21 FLOP/byte of issued loads.
// ---------------------------------------------------------------------------

typedef __bf16 bf16_t;
typedef __attribute__((ext_vector_type(16))) __bf16 v16bf;
typedef __attribute__((ext_vector_type(8)))  __bf16 bfx8;
typedef __attribute__((ext_vector_type(8)))  float  v8f;
typedef __attribute__((ext_vector_type(4)))  unsigned int u32x4;
typedef __attribute__((ext_vector_type(8)))  int i32x8;
typedef __attribute__((ext_vector_type(4)))  int i32x4;

#define MSDA_B       2
#define MSDA_NQ      10000
#define MSDA_NV      21760
#define MSDA_EMBED   256
#define MSDA_LEVELS  4
#define MSDA_POINTS  4

#if defined(__has_builtin)
#if __has_builtin(__builtin_amdgcn_tensor_load_to_lds) && \
    __has_builtin(__builtin_amdgcn_s_wait_tensorcnt)
#define MSDA_USE_TDM 1
#endif
#endif
#ifndef MSDA_USE_TDM
#define MSDA_USE_TDM 0
#endif

static __device__ __forceinline__ uint16_t f32_to_bf16_rne(float f) {
    union { float f; uint32_t u; } c; c.f = f;
    uint32_t u = c.u;
    uint32_t r = u + 0x7FFFu + ((u >> 16) & 1u);   // round to nearest even
    return (uint16_t)(r >> 16);
}

// --------------------------- conversion kernels ----------------------------

__global__ __launch_bounds__(256)
void msda_cvt_bf16(const float* __restrict__ in, uint16_t* __restrict__ out, int n) {
    int i = blockIdx.x * 256 + threadIdx.x;
    if (i < n) out[i] = f32_to_bf16_rne(in[i]);
}

// W: [K][N] row-major f32  ->  Wt: [N][K] row-major bf16
__global__ __launch_bounds__(256)
void msda_transpose_bf16(const float* __restrict__ W, uint16_t* __restrict__ Wt,
                         int K, int N) {
    int i = blockIdx.x * 256 + threadIdx.x;
    if (i < K * N) {
        int k = i / N, n = i % N;
        Wt[(size_t)n * K + k] = f32_to_bf16_rne(W[i]);
    }
}

// ------------------------------ WMMA GEMM ----------------------------------
// C[M][N] = A[M][K](bf16) * Bt[N][K](bf16)^T + bias[N]   (f32 accumulate/out)
// Fragment layouts used here (wave32; A/B K order is contiguous, not the ISA table):
//   A 16x32 bf16: lane(half=l>>4, m=l&15) holds K = half*16+[0..15]
//   B 32x16 bf16: lane(half, n=l&15)      holds K = half*16+[0..15] of column n
//   C/D 16x16 f32: lane(half, n=l&15), vgpr r -> row m = r + half*8, col n

union FragBF { v16bf v; bfx8 h[2]; };

#define MSDA_WMMA(ACC, AF, BF)                                                  \
    ACC = __builtin_amdgcn_wmma_f32_16x16x32_bf16(false, (AF).v, false, (BF).v, \
                                                  (short)0, ACC, false, false)

__global__ __launch_bounds__(256)
void msda_gemm_bf16(const uint16_t* __restrict__ A,   // [M][K] bf16
                    const uint16_t* __restrict__ Bt,  // [N][K] bf16
                    const float*   __restrict__ bias, // [N]
                    float*         __restrict__ C,    // [M][N] f32
                    int M, int N, int K) {
    extern __shared__ uint16_t shB[];                 // 64 rows x 256 K = 32 KB

    const int lane = threadIdx.x & 31;
    const int wave = threadIdx.x >> 5;
    const int tn   = blockIdx.y;                      // 64-wide N strip
    const int half = lane >> 4;
    const int mr   = lane & 15;

    // ---- stage the 64x256 bf16 weight panel for this N strip into LDS ----
    const uint16_t* panel = Bt + (size_t)tn * 64 * 256;
#if MSDA_USE_TDM
    if (wave == 0) {
        // Tensor DMA descriptor (CDNA5 ISA 8.3/8.4): 2D tile, 2-byte elems,
        // tile = tensor = 64 rows x 256 elems, row stride 256, lds_addr = 0
        // (dynamic LDS base).
        const uint64_t ga = (uint64_t)(uintptr_t)panel;
        u32x4 g0;
        g0[0] = 1u;                                   // count=1, user mode
        g0[1] = 0u;                                   // lds_addr (dyn-LDS base)
        g0[2] = (uint32_t)ga;                         // global_addr[31:0]
        g0[3] = (uint32_t)((ga >> 32) & 0x01FFFFFFu)  // global_addr[56:32]
              | (2u << 30);                           // type = 2 (image)
        i32x8 g1;
        g1[0] = (1 << 16);                            // data_size = 1 (2 bytes)
        g1[1] = (256 << 16);                          // tensor_dim0 = 256
        g1[2] = (64 << 16);                           // tensor_dim1 = 64
        g1[3] = (256 << 16);                          // tile_dim0 = 256
        g1[4] = 64;                                   // tile_dim1 = 64
        g1[5] = 256;                                  // tensor_dim0_stride = 256
        g1[6] = 0;
        g1[7] = 0;
        i32x4 g2 = {0, 0, 0, 0};
        i32x4 g3 = {0, 0, 0, 0};
#if __clang_major__ >= 23
        i32x8 g4 = {0, 0, 0, 0, 0, 0, 0, 0};
        __builtin_amdgcn_tensor_load_to_lds(g0, g1, g2, g3, g4, 0);
#else
        __builtin_amdgcn_tensor_load_to_lds(g0, g1, g2, g3, 0);
#endif
        __builtin_amdgcn_s_wait_tensorcnt(0);
    }
#else
    // Cooperative fallback: 2048 x 16B chunks, 8 per thread.
    {
        const uint32_t* src = (const uint32_t*)panel;
        uint32_t*       dst = (uint32_t*)shB;
        #pragma unroll
        for (int t = 0; t < 8; ++t) {
            int c = threadIdx.x + t * 256;            // 16B chunk index
            ((uint4*)dst)[c] = ((const uint4*)src)[c];
        }
    }
#endif
    __syncthreads();

    // ---- per-wave 32(M) x 64(N) tile ----
    const int rowBase = blockIdx.x * 256 + wave * 32;
    if (rowBase < M) {                                // M is a multiple of 32
        const bf16_t* aRow0 = (const bf16_t*)A + (size_t)(rowBase +      mr) * K;
        const bf16_t* aRow1 = (const bf16_t*)A + (size_t)(rowBase + 16 + mr) * K;
        const bf16_t* sB0 = (const bf16_t*)shB + (size_t)( 0 + mr) * 256;
        const bf16_t* sB1 = (const bf16_t*)shB + (size_t)(16 + mr) * 256;
        const bf16_t* sB2 = (const bf16_t*)shB + (size_t)(32 + mr) * 256;
        const bf16_t* sB3 = (const bf16_t*)shB + (size_t)(48 + mr) * 256;

        v8f acc00 = {}, acc01 = {}, acc02 = {}, acc03 = {};
        v8f acc10 = {}, acc11 = {}, acc12 = {}, acc13 = {};

        #pragma unroll
        for (int k = 0; k < 256; k += 32) {           // K == 256 always here
            if (k + 64 < K) {
                __builtin_prefetch(aRow0 + k + 64, 0, 3);
                __builtin_prefetch(aRow1 + k + 64, 0, 3);
            }
            FragBF a0, a1, b0, b1, b2, b3;
            a0.h[0] = *(const bfx8*)(aRow0 + k     + half * 16);
            a0.h[1] = *(const bfx8*)(aRow0 + k + 8 + half * 16);
            a1.h[0] = *(const bfx8*)(aRow1 + k     + half * 16);
            a1.h[1] = *(const bfx8*)(aRow1 + k + 8 + half * 16);
            const int kb = k + half * 16;
            b0.h[0] = *(const bfx8*)(sB0 + kb); b0.h[1] = *(const bfx8*)(sB0 + kb + 8);
            b1.h[0] = *(const bfx8*)(sB1 + kb); b1.h[1] = *(const bfx8*)(sB1 + kb + 8);
            b2.h[0] = *(const bfx8*)(sB2 + kb); b2.h[1] = *(const bfx8*)(sB2 + kb + 8);
            b3.h[0] = *(const bfx8*)(sB3 + kb); b3.h[1] = *(const bfx8*)(sB3 + kb + 8);

            MSDA_WMMA(acc00, a0, b0);  MSDA_WMMA(acc10, a1, b0);
            MSDA_WMMA(acc01, a0, b1);  MSDA_WMMA(acc11, a1, b1);
            MSDA_WMMA(acc02, a0, b2);  MSDA_WMMA(acc12, a1, b2);
            MSDA_WMMA(acc03, a0, b3);  MSDA_WMMA(acc13, a1, b3);
        }

        const float bb0 = bias[tn * 64 +  0 + mr];
        const float bb1 = bias[tn * 64 + 16 + mr];
        const float bb2 = bias[tn * 64 + 32 + mr];
        const float bb3 = bias[tn * 64 + 48 + mr];

        #pragma unroll
        for (int r = 0; r < 8; ++r) {
            const int r0 = rowBase +      half * 8 + r;
            const int r1 = rowBase + 16 + half * 8 + r;
            float* c0 = C + (size_t)r0 * N + tn * 64 + mr;
            float* c1 = C + (size_t)r1 * N + tn * 64 + mr;
            c0[ 0] = acc00[r] + bb0;  c1[ 0] = acc10[r] + bb0;
            c0[16] = acc01[r] + bb1;  c1[16] = acc11[r] + bb1;
            c0[32] = acc02[r] + bb2;  c1[32] = acc12[r] + bb2;
            c0[48] = acc03[r] + bb3;  c1[48] = acc13[r] + bb3;
        }
    }
}

// --------------------------- sampling kernel -------------------------------
// One block per (b,q); wave = head, lane = channel d (0..31).
// v_proj layout: [B][NV][H*32] f32 -> per-corner fetch is 128B contiguous.

static __device__ __forceinline__ float msda_fetch(const float* __restrict__ base,
                                                   int x, int y, int Ws, int Hs) {
    if (x < 0 || x >= Ws || y < 0 || y >= Hs) return 0.0f;
    return base[(size_t)(y * Ws + x) * MSDA_EMBED];
}

__global__ __launch_bounds__(256)
void msda_sample(const float* __restrict__ vproj,   // [B*NV][256]
                 const float* __restrict__ off,     // [B*NQ][256]
                 const float* __restrict__ attn,    // [B*NQ][128]
                 float*       __restrict__ outI,    // [B*NQ][256]
                 uint16_t*    __restrict__ outIbf)  // [B*NQ][256] bf16
{
    const int bq = blockIdx.x;
    const int h  = threadIdx.x >> 5;
    const int d  = threadIdx.x & 31;
    const int b  = bq / MSDA_NQ;

    static const int lvlH[4]     = {128, 64, 32, 16};
    static const int lvlW[4]     = {128, 64, 32, 16};
    static const int lvlStart[4] = {0, 16384, 20480, 21504};

    const float* offRow = off  + (size_t)bq * 256 + h * 32;  // (l,p,2)
    const float* attRow = attn + (size_t)bq * 128 + h * 16;  // (l,p)

    float acc = 0.0f;
    #pragma unroll
    for (int l = 0; l < MSDA_LEVELS; ++l) {
        float a0 = attRow[l * 4 + 0], a1 = attRow[l * 4 + 1];
        float a2 = attRow[l * 4 + 2], a3 = attRow[l * 4 + 3];
        float mx = fmaxf(fmaxf(a0, a1), fmaxf(a2, a3));
        float e[4];
        e[0] = __expf(a0 - mx); e[1] = __expf(a1 - mx);
        e[2] = __expf(a2 - mx); e[3] = __expf(a3 - mx);
        const float inv = 1.0f / (e[0] + e[1] + e[2] + e[3]);

        const int Hs = lvlH[l], Ws = lvlW[l];
        const float* base =
            vproj + ((size_t)b * MSDA_NV + lvlStart[l]) * MSDA_EMBED + h * 32 + d;

        #pragma unroll
        for (int p = 0; p < MSDA_POINTS; ++p) {
            const float w  = e[p] * inv;
            const float gx = offRow[l * 8 + p * 2 + 0];
            const float gy = offRow[l * 8 + p * 2 + 1];
            const float px = (gx + 1.0f) * 0.5f * (float)(Ws - 1);
            const float py = (gy + 1.0f) * 0.5f * (float)(Hs - 1);
            const float fx = floorf(px), fy = floorf(py);
            const float wx = px - fx,   wy = py - fy;
            const int x0 = (int)fx, y0 = (int)fy;

            const float v00 = msda_fetch(base, x0,     y0,     Ws, Hs);
            const float v01 = msda_fetch(base, x0 + 1, y0,     Ws, Hs);
            const float v10 = msda_fetch(base, x0,     y0 + 1, Ws, Hs);
            const float v11 = msda_fetch(base, x0 + 1, y0 + 1, Ws, Hs);

            const float top = v00 + wx * (v01 - v00);
            const float bot = v10 + wx * (v11 - v10);
            acc += w * (top + wy * (bot - top));
        }
    }

    const size_t oi = (size_t)bq * 256 + h * 32 + d;
    outI[oi]   = acc;
    outIbf[oi] = f32_to_bf16_rne(acc);
}

// ------------------------------- launcher ----------------------------------

static inline void launch_gemm(const uint16_t* A, const uint16_t* Bt,
                               const float* bias, float* C,
                               int M, int N, hipStream_t stream) {
    dim3 grid((M + 255) / 256, N / 64);
    msda_gemm_bf16<<<grid, 256, 64 * 256 * sizeof(uint16_t), stream>>>(
        A, Bt, bias, C, M, N, 256);
}

extern "C" void kernel_launch(void* const* d_in, const int* in_sizes, int n_in,
                              void* d_out, int out_size, void* d_ws, size_t ws_size,
                              hipStream_t stream) {
    (void)in_sizes; (void)n_in; (void)out_size; (void)ws_size;

    const float* query  = (const float*)d_in[0];
    /* d_in[1] = key (unused by reference) */
    const float* value  = (const float*)d_in[2];
    const float* W_off  = (const float*)d_in[3];
    const float* b_off  = (const float*)d_in[4];
    const float* W_attn = (const float*)d_in[5];
    const float* b_attn = (const float*)d_in[6];
    const float* W_v    = (const float*)d_in[7];
    const float* b_v    = (const float*)d_in[8];
    const float* W_out  = (const float*)d_in[9];
    const float* b_out  = (const float*)d_in[10];
    float* out = (float*)d_out;

    const size_t MQ = (size_t)MSDA_B * MSDA_NQ;        // 20000 query rows
    const size_t MV = (size_t)MSDA_B * MSDA_NV;        // 43520 value rows

    char*  ws  = (char*)d_ws;
    size_t off_b = 0;
    auto walloc = [&](size_t bytes) -> char* {
        char* p = ws + off_b;
        off_b = (off_b + bytes + 255) & ~(size_t)255;
        return p;
    };

    uint16_t* q_bf    = (uint16_t*)walloc(MQ * MSDA_EMBED * 2);
    uint16_t* v_bf    = (uint16_t*)walloc(MV * MSDA_EMBED * 2);
    uint16_t* Wv_t    = (uint16_t*)walloc(256 * 256 * 2);
    uint16_t* Woff_t  = (uint16_t*)walloc(256 * 256 * 2);
    uint16_t* Wattn_t = (uint16_t*)walloc(128 * 256 * 2);
    uint16_t* Wout_t  = (uint16_t*)walloc(256 * 256 * 2);
    float*    v_proj  = (float*)   walloc(MV * MSDA_EMBED * 4);
    float*    offBuf  = (float*)   walloc(MQ * 256 * 4);
    float*    attBuf  = (float*)   walloc(MQ * 128 * 4);
    float*    outI    = (float*)   walloc(MQ * 256 * 4);
    uint16_t* outIbf  = (uint16_t*)walloc(MQ * 256 * 2);

    // 1) activations -> bf16
    {
        int nq = (int)(MQ * MSDA_EMBED);
        int nv = (int)(MV * MSDA_EMBED);
        msda_cvt_bf16<<<(nq + 255) / 256, 256, 0, stream>>>(query, q_bf, nq);
        msda_cvt_bf16<<<(nv + 255) / 256, 256, 0, stream>>>(value, v_bf, nv);
    }
    // 2) weights -> transposed bf16
    msda_transpose_bf16<<<(256 * 256 + 255) / 256, 256, 0, stream>>>(W_v,    Wv_t,    256, 256);
    msda_transpose_bf16<<<(256 * 256 + 255) / 256, 256, 0, stream>>>(W_off,  Woff_t,  256, 256);
    msda_transpose_bf16<<<(256 * 128 + 255) / 256, 256, 0, stream>>>(W_attn, Wattn_t, 256, 128);
    msda_transpose_bf16<<<(256 * 256 + 255) / 256, 256, 0, stream>>>(W_out,  Wout_t,  256, 256);

    // 3) projections
    launch_gemm(v_bf,   Wv_t,    b_v,    v_proj, (int)MV, 256, stream);
    launch_gemm(q_bf,   Woff_t,  b_off,  offBuf, (int)MQ, 256, stream);
    launch_gemm(q_bf,   Wattn_t, b_attn, attBuf, (int)MQ, 128, stream);

    // 4) softmax + bilinear sampling + head concat
    msda_sample<<<(int)MQ, 256, 0, stream>>>(v_proj, offBuf, attBuf, outI, outIbf);

    // 5) output projection -> d_out
    launch_gemm(outIbf, Wout_t,  b_out,  out,    (int)MQ, 256, stream);
}
